// GraphNetBlock_55508157333731
// MI455X (gfx1250) — matrix operand order repaired and checked by reading, not compiled, not run on hardware
//
#include <hip/hip_runtime.h>

#define D 128

typedef __attribute__((ext_vector_type(16))) __bf16        v16bf;
typedef __attribute__((ext_vector_type(2)))  __bf16        v2bf;
typedef __attribute__((ext_vector_type(8)))  float         v8f;
typedef __attribute__((ext_vector_type(8)))  unsigned int  v8u;

union BF16x16 {
    v16bf          v;
    unsigned short u[16];
    unsigned int   d[8];
    v8u            w;
};

// f32 -> bf16 via native cast; backend picks best lowering for gfx1250
__device__ __forceinline__ unsigned short f2bf(float f) {
    __bf16 h = (__bf16)f;
    return __builtin_bit_cast(unsigned short, h);
}

// pack two f32 -> one dword holding two bf16 (lo, hi)
__device__ __forceinline__ unsigned int pk2bf(float lo, float hi) {
#if __has_builtin(__builtin_amdgcn_cvt_pk_bf16_f32)
    v2bf r = __builtin_amdgcn_cvt_pk_bf16_f32(lo, hi);
    return __builtin_bit_cast(unsigned int, r);
#else
    return (unsigned int)f2bf(lo) | ((unsigned int)f2bf(hi) << 16);
#endif
}

// convert 8 consecutive f32 -> 4 packed bf16 dwords
__device__ __forceinline__ void cvt8(const float* __restrict__ p, unsigned int* dst) {
    float4 a = *(const float4*)p;
    float4 b = *(const float4*)(p + 4);
    dst[0] = pk2bf(a.x, a.y);
    dst[1] = pk2bf(a.z, a.w);
    dst[2] = pk2bf(b.x, b.y);
    dst[3] = pk2bf(b.z, b.w);
}

__device__ __forceinline__ void ld8(const unsigned short* p, unsigned short* dst) {
    *(uint4*)dst = *(const uint4*)p;
}

// ---------------------------------------------------------------------------
// Pack W[K x 128] (row-major f32) into bf16 B-operand fragments.
// ---------------------------------------------------------------------------
__global__ void pack_w_kernel(const float* __restrict__ W,
                              unsigned short* __restrict__ out, int K) {
    int idx   = blockIdx.x * 256 + threadIdx.x;
    int total = (K >> 5) * 8 * 32 * 16;
    if (idx >= total) return;
    int e    = idx & 15;
    int lane = (idx >> 4) & 31;
    int nt   = (idx >> 9) & 7;
    int ks   = idx >> 12;
    int g    = lane >> 4;
    int n    = nt * 16 + (lane & 15);
    int k    = ks * 32 + g * 8 + e + ((e >> 3) << 3);
    out[idx] = f2bf(W[(long)k * D + n]);
}

// elementwise f32 -> bf16 (4 elems / thread)
__global__ void cvt_f32_bf16_kernel(const float* __restrict__ in,
                                    unsigned short* __restrict__ out, long n4) {
    long i = (long)blockIdx.x * 256 + threadIdx.x;
    if (i >= n4) return;
    float4 v = ((const float4*)in)[i];
    uint2 o;
    o.x = pk2bf(v.x, v.y);
    o.y = pk2bf(v.z, v.w);
    ((uint2*)out)[i] = o;
}

// ---------------------------------------------------------------------------
// Software-pipelined GEMM core: KS k-steps x 8 n-tiles, NT row-tiles.
// B fragments: depth-4 register ring (prefetch step+3).
// A fragments: ping-pong, next k-step loaded at nt==4.
// ---------------------------------------------------------------------------
template <int KS, int NT, class ALoad>
__device__ __forceinline__ void gemm_pipe(const unsigned short* __restrict__ Wp,
                                          int lane, v8f (*acc)[8], ALoad loadA) {
    const v8u* wp = (const v8u*)Wp;
    BF16x16 a[2][NT], b[4];
    loadA(0, a[0]);
    b[0].w = wp[0 * 32 + lane];
    b[1].w = wp[1 * 32 + lane];
    b[2].w = wp[2 * 32 + lane];
#pragma unroll
    for (int step = 0; step < KS * 8; ++step) {
        const int ks = step >> 3, nt = step & 7;
        if (step + 3 < KS * 8) b[(step + 3) & 3].w = wp[(step + 3) * 32 + lane];
        if (nt == 4 && ks + 1 < KS) loadA(ks + 1, a[(ks + 1) & 1]);
#pragma unroll
        for (int t = 0; t < NT; ++t)
            acc[t][nt] = __builtin_amdgcn_wmma_f32_16x16x32_bf16(
                false, a[ks & 1][t].v, false, b[step & 3].v, (short)0,
                acc[t][nt], false, false);
    }
}

template <int NT>
__device__ __forceinline__ void gemm_lds(const unsigned short* __restrict__ hbuf,
                                         const unsigned short* __restrict__ Wp,
                                         int lane, v8f (*acc)[8]) {
    const int g = lane >> 4, l15 = lane & 15;
    auto loadA = [&](int ks, BF16x16* a) {
#pragma unroll
        for (int t = 0; t < NT; ++t) {
            const unsigned short* hrow = hbuf + (t * 16 + l15) * D;
            ld8(hrow + ks * 32 + 8 * g,      &a[t].u[0]);
            ld8(hrow + ks * 32 + 16 + 8 * g, &a[t].u[8]);
        }
    };
    gemm_pipe<4, NT>(Wp, lane, acc, loadA);
}

template <int NT>
__device__ __forceinline__ void bias_relu_to_lds(v8f (*acc)[8], const float* __restrict__ b,
                                                 unsigned short* hbuf, int lane) {
    int g = lane >> 4, l15 = lane & 15;
#pragma unroll
    for (int nt = 0; nt < 8; ++nt) {
        float bias = b[nt * 16 + l15];
#pragma unroll
        for (int t = 0; t < NT; ++t)
#pragma unroll
            for (int i = 0; i < 8; ++i) {
                float h = acc[t][nt][i] + bias;
                h = h > 0.0f ? h : 0.0f;
                hbuf[(t * 16 + g * 8 + i) * D + nt * 16 + l15] = f2bf(h);
            }
    }
    asm volatile("s_wait_dscnt 0" ::: "memory");
}

__device__ __forceinline__ void bias_ln_stats(v8f* acc, const float* __restrict__ b,
                                              int l15, float* mu, float* rv) {
    float s[8], q[8];
#pragma unroll
    for (int i = 0; i < 8; ++i) { s[i] = 0.0f; q[i] = 0.0f; }
#pragma unroll
    for (int nt = 0; nt < 8; ++nt) {
        float bias = b[nt * 16 + l15];
#pragma unroll
        for (int i = 0; i < 8; ++i) {
            float x = acc[nt][i] + bias;
            acc[nt][i] = x;
            s[i] += x;
            q[i] += x * x;
        }
    }
#pragma unroll
    for (int m = 1; m < 16; m <<= 1) {
#pragma unroll
        for (int i = 0; i < 8; ++i) {
            s[i] += __shfl_xor(s[i], m, 32);
            q[i] += __shfl_xor(q[i], m, 32);
        }
    }
#pragma unroll
    for (int i = 0; i < 8; ++i) {
        float m   = s[i] * (1.0f / (float)D);
        float var = q[i] * (1.0f / (float)D) - m * m;
        mu[i] = m;
        rv[i] = rsqrtf(var + 1e-5f);
    }
}

// ---------------------------------------------------------------------------
// Edge update: 4 waves/block, 32 edges per wave (2 A tiles share each B frag).
// ---------------------------------------------------------------------------
template <bool PRE>
__global__ __launch_bounds__(128)
void edge_kernel(const float* __restrict__ nodef, const float* __restrict__ edgef,
                 const unsigned short* __restrict__ nodeb,
                 const unsigned short* __restrict__ edgeb,
                 const int* __restrict__ senders, const int* __restrict__ receivers,
                 const unsigned short* __restrict__ W1p,
                 const unsigned short* __restrict__ W2p,
                 const unsigned short* __restrict__ W3p,
                 const float* __restrict__ b1, const float* __restrict__ b2,
                 const float* __restrict__ b3, const float* __restrict__ gamma,
                 const float* __restrict__ beta,
                 float* __restrict__ agg, float* __restrict__ out_edge,
                 int n_edges) {
    __shared__ unsigned short ldsH[4 * 32 * D];          // 32 KB
    const int lane = threadIdx.x & 31;
    const int wave = threadIdx.x >> 5;
    const int tile = blockIdx.x * 4 + wave;
    const int e0   = tile * 32;
    if (e0 >= n_edges) return;
    const int g   = lane >> 4;
    const int l15 = lane & 15;

    int sidx[2], ridx[2], erow[2];
#pragma unroll
    for (int t = 0; t < 2; ++t) {
        int r = e0 + t * 16 + l15;
        if (r >= n_edges) r = n_edges - 1;
        erow[t] = r;
        sidx[t] = senders[r];
        ridx[t] = receivers[r];
    }
    unsigned short* hbuf = ldsH + wave * 32 * D;

    v8f acc[2][8] = {};
    // layer 1: [32 x 384] @ We1 (pipelined)
    auto loadA1 = [&](int ks, BF16x16* a) {
        int k0 = ks * 32, kl = k0 & 127;
#pragma unroll
        for (int t = 0; t < 2; ++t) {
            if (PRE) {
                const unsigned short* src =
                    (k0 < 128) ? nodeb + (long)sidx[t] * D
                  : (k0 < 256) ? nodeb + (long)ridx[t] * D
                               : edgeb + (long)erow[t] * D;
                ld8(src + kl + 8 * g,      &a[t].u[0]);
                ld8(src + kl + 16 + 8 * g, &a[t].u[8]);
            } else {
                const float* src =
                    (k0 < 128) ? nodef + (long)sidx[t] * D
                  : (k0 < 256) ? nodef + (long)ridx[t] * D
                               : edgef + (long)erow[t] * D;
                cvt8(src + kl + 8 * g,      &a[t].d[0]);
                cvt8(src + kl + 16 + 8 * g, &a[t].d[4]);
            }
        }
    };
    gemm_pipe<12, 2>(W1p, lane, acc, loadA1);
    bias_relu_to_lds<2>(acc, b1, hbuf, lane);

    v8f acc2[2][8] = {};
    gemm_lds<2>(hbuf, W2p, lane, acc2);                  // layer 2
    bias_relu_to_lds<2>(acc2, b2, hbuf, lane);

    v8f acc3[2][8] = {};
    gemm_lds<2>(hbuf, W3p, lane, acc3);                  // layer 3

    float mu[2][8], rv[2][8];
#pragma unroll
    for (int t = 0; t < 2; ++t)
        bias_ln_stats(acc3[t], b3, l15, mu[t], rv[t]);

    int rcv[2][8];
#pragma unroll
    for (int t = 0; t < 2; ++t)
#pragma unroll
        for (int i = 0; i < 8; ++i) {
            int r = e0 + t * 16 + g * 8 + i;
            rcv[t][i] = receivers[r < n_edges ? r : (n_edges - 1)];
        }

#pragma unroll
    for (int nt = 0; nt < 8; ++nt) {
        int   n  = nt * 16 + l15;
        float gm = gamma[n], bt = beta[n];
#pragma unroll
        for (int t = 0; t < 2; ++t)
#pragma unroll
            for (int i = 0; i < 8; ++i) {
                int er = e0 + t * 16 + g * 8 + i;
                if (er < n_edges) {
                    float y = (acc3[t][nt][i] - mu[t][i]) * rv[t][i] * gm + bt;
                    atomicAdd(&agg[(long)rcv[t][i] * D + n], y);   // pre-residual
                    out_edge[(long)er * D + n] = y + edgef[(long)er * D + n];
                }
            }
    }
}

// ---------------------------------------------------------------------------
// Node update: input = [node_features | agg]  (K = 256), 16 rows per wave.
// ---------------------------------------------------------------------------
template <bool PRE>
__global__ __launch_bounds__(128)
void node_kernel(const float* __restrict__ nodef, const unsigned short* __restrict__ nodeb,
                 const float* __restrict__ agg,
                 const unsigned short* __restrict__ W1p,
                 const unsigned short* __restrict__ W2p,
                 const unsigned short* __restrict__ W3p,
                 const float* __restrict__ b1, const float* __restrict__ b2,
                 const float* __restrict__ b3, const float* __restrict__ gamma,
                 const float* __restrict__ beta,
                 float* __restrict__ out_node, int n_nodes) {
    __shared__ unsigned short ldsH[4 * 16 * D];
    const int lane = threadIdx.x & 31;
    const int wave = threadIdx.x >> 5;
    const int tile = blockIdx.x * 4 + wave;
    const int n0   = tile * 16;
    if (n0 >= n_nodes) return;
    const int g   = lane >> 4;
    const int l15 = lane & 15;
    int myrow = n0 + l15;
    if (myrow >= n_nodes) myrow = n_nodes - 1;

    const float*          srcN  = nodef + (long)myrow * D;
    const unsigned short* srcNb = PRE ? nodeb + (long)myrow * D : nullptr;
    const float*          srcA  = agg + (long)myrow * D;
    unsigned short* hbuf = ldsH + wave * 16 * D;

    v8f acc[1][8] = {};
    auto loadA1 = [&](int ks, BF16x16* a) {
        int k0 = ks * 32, kl = k0 & 127;
        if (PRE && k0 < 128) {
            ld8(srcNb + kl + 8 * g,      &a[0].u[0]);
            ld8(srcNb + kl + 16 + 8 * g, &a[0].u[8]);
        } else {
            const float* src = (k0 < 128) ? srcN : srcA;
            cvt8(src + kl + 8 * g,      &a[0].d[0]);
            cvt8(src + kl + 16 + 8 * g, &a[0].d[4]);
        }
    };
    gemm_pipe<8, 1>(W1p, lane, acc, loadA1);
    bias_relu_to_lds<1>(acc, b1, hbuf, lane);

    v8f acc2[1][8] = {};
    gemm_lds<1>(hbuf, W2p, lane, acc2);
    bias_relu_to_lds<1>(acc2, b2, hbuf, lane);

    v8f acc3[1][8] = {};
    gemm_lds<1>(hbuf, W3p, lane, acc3);

    float mu[8], rv[8];
    bias_ln_stats(acc3[0], b3, l15, mu, rv);

#pragma unroll
    for (int nt = 0; nt < 8; ++nt) {
        int   n  = nt * 16 + l15;
        float gm = gamma[n], bt = beta[n];
#pragma unroll
        for (int i = 0; i < 8; ++i) {
            int nr = n0 + g * 8 + i;
            if (nr < n_nodes) {
                float y = (acc3[0][nt][i] - mu[i]) * rv[i] * gm + bt;
                out_node[(long)nr * D + n] = y + nodef[(long)nr * D + n];
            }
        }
    }
}

// ---------------------------------------------------------------------------
extern "C" void kernel_launch(void* const* d_in, const int* in_sizes, int n_in,
                              void* d_out, int out_size, void* d_ws, size_t ws_size,
                              hipStream_t stream) {
    const float* nodef     = (const float*)d_in[0];
    const float* edgef     = (const float*)d_in[1];
    const int*   senders   = (const int*)d_in[2];
    const int*   receivers = (const int*)d_in[3];
    const float* We1 = (const float*)d_in[4];  const float* be1  = (const float*)d_in[5];
    const float* We2 = (const float*)d_in[6];  const float* be2  = (const float*)d_in[7];
    const float* We3 = (const float*)d_in[8];  const float* be3  = (const float*)d_in[9];
    const float* ge  = (const float*)d_in[10]; const float* bege = (const float*)d_in[11];
    const float* Wn1 = (const float*)d_in[12]; const float* bn1  = (const float*)d_in[13];
    const float* Wn2 = (const float*)d_in[14]; const float* bn2  = (const float*)d_in[15];
    const float* Wn3 = (const float*)d_in[16]; const float* bn3  = (const float*)d_in[17];
    const float* gn  = (const float*)d_in[18]; const float* begn = (const float*)d_in[19];

    const int n_nodes = in_sizes[0] / D;
    const int n_edges = in_sizes[1] / D;

    char*  ws  = (char*)d_ws;
    float* agg = (float*)ws;
    size_t off = ((size_t)n_nodes * D * sizeof(float) + 1023) & ~(size_t)1023;
    unsigned short* We1p = (unsigned short*)(ws + off); off += (size_t)3 * D * D * 2;
    unsigned short* We2p = (unsigned short*)(ws + off); off += (size_t)D * D * 2;
    unsigned short* We3p = (unsigned short*)(ws + off); off += (size_t)D * D * 2;
    unsigned short* Wn1p = (unsigned short*)(ws + off); off += (size_t)2 * D * D * 2;
    unsigned short* Wn2p = (unsigned short*)(ws + off); off += (size_t)D * D * 2;
    unsigned short* Wn3p = (unsigned short*)(ws + off); off += (size_t)D * D * 2;
    off = (off + 1023) & ~(size_t)1023;
    unsigned short* nodeb = (unsigned short*)(ws + off); off += (size_t)n_nodes * D * 2;
    unsigned short* edgeb = (unsigned short*)(ws + off); off += (size_t)n_edges * D * 2;
    const bool pre = (ws_size >= off);

    auto pack = [&](const float* W, unsigned short* o, int K) {
        int total = K * D;
        pack_w_kernel<<<(total + 255) / 256, 256, 0, stream>>>(W, o, K);
    };
    pack(We1, We1p, 3 * D);
    pack(We2, We2p, D);
    pack(We3, We3p, D);
    pack(Wn1, Wn1p, 2 * D);
    pack(Wn2, Wn2p, D);
    pack(Wn3, Wn3p, D);

    if (pre) {
        long n4n = (long)n_nodes * D / 4;
        long n4e = (long)n_edges * D / 4;
        cvt_f32_bf16_kernel<<<(int)((n4n + 255) / 256), 256, 0, stream>>>(nodef, nodeb, n4n);
        cvt_f32_bf16_kernel<<<(int)((n4e + 255) / 256), 256, 0, stream>>>(edgef, edgeb, n4e);
    }

    hipMemsetAsync(agg, 0, (size_t)n_nodes * D * sizeof(float), stream);

    float* out_node = (float*)d_out;
    float* out_edge = out_node + (size_t)n_nodes * D;

    int eblocks = ((n_edges + 31) / 32 + 3) / 4;
    int nblocks = ((n_nodes + 15) / 16 + 3) / 4;
    if (pre) {
        edge_kernel<true><<<eblocks, 128, 0, stream>>>(
            nodef, edgef, nodeb, edgeb, senders, receivers, We1p, We2p, We3p,
            be1, be2, be3, ge, bege, agg, out_edge, n_edges);
        node_kernel<true><<<nblocks, 128, 0, stream>>>(
            nodef, nodeb, agg, Wn1p, Wn2p, Wn3p,
            bn1, bn2, bn3, gn, begn, out_node, n_nodes);
    } else {
        edge_kernel<false><<<eblocks, 128, 0, stream>>>(
            nodef, edgef, nullptr, nullptr, senders, receivers, We1p, We2p, We3p,
            be1, be2, be3, ge, bege, agg, out_edge, n_edges);
        node_kernel<false><<<nblocks, 128, 0, stream>>>(
            nodef, nullptr, agg, Wn1p, Wn2p, Wn3p,
            bn1, bn2, bn3, gn, begn, out_node, n_nodes);
    }
}
